// CrossframeLocalInterpolationModule_39384850105054
// MI455X (gfx1250) — matrix operand order repaired and checked by reading, not compiled, not run on hardware
//
#include <hip/hip_runtime.h>
#include <hip/hip_bf16.h>

typedef __attribute__((ext_vector_type(16))) _Float16 v16h;
typedef __attribute__((ext_vector_type(8)))  float    v8f;

#define CDIM 32
#define KNBR 9
#define ROWP 68   // padded cat-row stride in halves (64 data + 4 pad) to spread LDS banks

__global__ __launch_bounds__(256)
void clim_kernel(const float* __restrict__ lv,
                 const float* __restrict__ hidden,
                 const float* __restrict__ Wm,
                 const float* __restrict__ b_lin,
                 const float* __restrict__ b_aflow,
                 const float* __restrict__ alpha_p,
                 const float* __restrict__ beta_p,
                 const int*   __restrict__ nidx,
                 float* __restrict__ out,
                 int N)
{
    // 8 waves/block, each wave owns a private 16x(64+pad) f16 cat tile: ~17.4 KB
    __shared__ _Float16 lds_cat[8 * 16 * ROWP];

    const int lane = threadIdx.x & 31;
    const int wv   = threadIdx.x >> 5;
    const int base = (blockIdx.x * 8 + wv) * 16;   // first vertex of this wave's tile
    _Float16* my = lds_cat + wv * (16 * ROWP);

    const float alpha = alpha_p[0];
    const float beta  = beta_p[0];
    const int hi  = lane >> 4;    // half-wave id
    const int col = lane & 15;    // B/D column within 16-wide tile

    // ---- B fragments: W (64x32 f32, row-major) -> f16, contiguous-k B order (same as A):
    // lane n holds column N=n&15; halves h=0..15 hold K = 16*(n>>4) + h.
    v16h b00, b01, b10, b11;      // [kk-step 0/1][N-tile 0/1]
    #pragma unroll
    for (int h = 0; h < 16; ++h) {
        int Kb = hi * 16 + h;
        b00[h] = (_Float16)Wm[(Kb)      * CDIM + col];
        b01[h] = (_Float16)Wm[(Kb)      * CDIM + 16 + col];
        b10[h] = (_Float16)Wm[(Kb + 32) * CDIM + col];
        b11[h] = (_Float16)Wm[(Kb + 32) * CDIM + 16 + col];
    }

    const float b_af = b_aflow[lane];   // lane == channel

    // ---- Phase 1: gather + distance weights + aflow, one vertex at a time
    for (int v = 0; v < 16; ++v) {
        int n = base + v;
        _Float16 catA = (_Float16)0.f, catL = (_Float16)0.f;
        if (n < N) {                                   // wave-uniform guard
            float lvv = lv[n * CDIM + lane];
            int myidx = (lane < KNBR) ? nidx[n * KNBR + lane] : -1;

            float r[KNBR], dist[KNBR];
            bool  val[KNBR];
            float dsum = 0.f;
            #pragma unroll
            for (int k = 0; k < KNBR; ++k) {
                int ki = __shfl(myidx, k, 32);         // scalar per wave
                val[k] = (ki >= 0);
                float rv = 0.f;
                if (ki >= 0)                           // wave-uniform branch
                    rv = hidden[(long)ki * CDIM + lane];  // coalesced 128B row
                r[k] = rv;
                float df = rv - lvv;
                float s = df * df;                     // 32-lane butterfly allreduce
                s += __shfl_xor(s, 16, 32);
                s += __shfl_xor(s,  8, 32);
                s += __shfl_xor(s,  4, 32);
                s += __shfl_xor(s,  2, 32);
                s += __shfl_xor(s,  1, 32);
                float d = val[k] ? sqrtf(s) : 0.f;
                dist[k] = d;
                dsum += d;
            }
            float inv = 1.0f / dsum;                   // matches reference (no guard)
            float af = b_af;
            #pragma unroll
            for (int k = 0; k < KNBR; ++k) {
                float dn = dist[k] * inv;
                float w  = (alpha - fminf(dn, alpha)) * beta;
                w = val[k] ? w : 0.f;
                af = fmaf(w, r[k], af);
            }
            catA = (_Float16)af;
            catL = (_Float16)lvv;
        }
        my[v * ROWP + lane]      = catA;   // cat[:, 0:32]  = aflow
        my[v * ROWP + 32 + lane] = catL;   // cat[:, 32:64] = lv
    }

    // ---- Phase 2: A fragments, contiguous-k order matching B:
    // lane L holds row M=L&15; halves h -> K = 16*(L>>4) + h
    const int m = lane & 15;
    v16h a0, a1;
    #pragma unroll
    for (int h = 0; h < 16; ++h) {
        int Ka = hi * 16 + h;
        a0[h] = my[m * ROWP + Ka];         // K  0..31 of cat row
        a1[h] = my[m * ROWP + 32 + Ka];    // K 32..63 of cat row
    }

    v8f acc0 = {}; v8f acc1 = {};
    acc0 = __builtin_amdgcn_wmma_f32_16x16x32_f16(false, a0, false, b00, (short)0, acc0, false, false);
    acc0 = __builtin_amdgcn_wmma_f32_16x16x32_f16(false, a1, false, b10, (short)0, acc0, false, false);
    acc1 = __builtin_amdgcn_wmma_f32_16x16x32_f16(false, a0, false, b01, (short)0, acc1, false, false);
    acc1 = __builtin_amdgcn_wmma_f32_16x16x32_f16(false, a1, false, b11, (short)0, acc1, false, false);

    // ---- Bias + ReLU + store. C/D layout: lane L -> col L&15, rows r + 8*(L>>4)
    const float bl0 = b_lin[col];
    const float bl1 = b_lin[col + 16];
    #pragma unroll
    for (int rr = 0; rr < 8; ++rr) {
        int row = hi * 8 + rr;
        int n = base + row;
        if (n < N) {
            out[n * CDIM + col]      = fmaxf(acc0[rr] + bl0, 0.f);
            out[n * CDIM + 16 + col] = fmaxf(acc1[rr] + bl1, 0.f);
        }
    }
}

extern "C" void kernel_launch(void* const* d_in, const int* in_sizes, int n_in,
                              void* d_out, int out_size, void* d_ws, size_t ws_size,
                              hipStream_t stream) {
    const float* lv       = (const float*)d_in[0];
    const float* hidden   = (const float*)d_in[1];
    const float* Wm       = (const float*)d_in[2];
    const float* b_lin    = (const float*)d_in[3];
    const float* b_aflow  = (const float*)d_in[4];
    const float* alpha_p  = (const float*)d_in[5];
    const float* beta_p   = (const float*)d_in[6];
    const int*   nidx     = (const int*)d_in[7];
    float* out = (float*)d_out;

    const int N = in_sizes[0] / CDIM;           // 500000
    const int blocks = (N + 127) / 128;         // 8 waves x 16 vertices per block

    clim_kernel<<<blocks, 256, 0, stream>>>(lv, hidden, Wm, b_lin, b_aflow,
                                            alpha_p, beta_p, nidx, out, N);
}
